// TernaryLinear_4638564680453
// MI455X (gfx1250) — matrix operand order repaired and checked by reading, not compiled, not run on hardware
//
#include <hip/hip_runtime.h>
#include <math.h>

// Problem dims (from reference): x[8,2048,2048] f32, W[2048,2048] f32, log_scale[2048] f32
#define MDIM 16384   // B*S
#define NDIM 2048    // D_OUT
#define KDIM 2048    // D_IN
#define NW   (2048u * 2048u)
#define MEDIAN_RANK 2097151u   // 0-indexed lower-middle of 4194304

typedef __attribute__((ext_vector_type(16))) __bf16        v16bf;
typedef __attribute__((ext_vector_type(8)))  float         v8f;
typedef __attribute__((ext_vector_type(4)))  unsigned int  u32x4;
typedef __attribute__((ext_vector_type(2)))  unsigned int  u32x2;
typedef __attribute__((ext_vector_type(4)))  int           v4i;

#ifndef __has_builtin
#define __has_builtin(x) 0
#endif

#if __has_builtin(__builtin_amdgcn_global_load_async_to_lds_b128)
#define HAVE_ASYNC_LDS 1
typedef __attribute__((address_space(1))) v4i* as1_v4i_ptr;
typedef __attribute__((address_space(3))) v4i* as3_v4i_ptr;
#else
#define HAVE_ASYNC_LDS 0
#endif

#if HAVE_ASYNC_LDS
#if __has_builtin(__builtin_amdgcn_s_wait_asynccnt)
#define WAIT_ASYNC0() __builtin_amdgcn_s_wait_asynccnt(0)
#else
#define WAIT_ASYNC0() asm volatile("s_wait_asynccnt 0x0" ::: "memory")
#endif
#else
#define WAIT_ASYNC0() ((void)0)
#endif

__device__ __forceinline__ unsigned short f32_to_bf16_rne(float f) {
  unsigned int u = __float_as_uint(f);
  u += 0x7FFFu + ((u >> 16) & 1u);
  return (unsigned short)(u >> 16);
}

// ---------------- median pre-pass: radix select on bits of |w| ----------------

__global__ void k_zero(unsigned int* __restrict__ p, int n) {
  int i = blockIdx.x * blockDim.x + threadIdx.x;
  if (i < n) p[i] = 0u;
}

__global__ void k_hist_hi(const float* __restrict__ w, unsigned int* __restrict__ hist) {
  unsigned int i = blockIdx.x * blockDim.x + threadIdx.x;
  unsigned int stride = gridDim.x * blockDim.x;
  for (unsigned int idx = i; idx < NW; idx += stride) {
    unsigned int b = __float_as_uint(fabsf(w[idx]));
    atomicAdd(&hist[b >> 16], 1u);
  }
}

__global__ void k_sel_hi(const unsigned int* __restrict__ hist, unsigned int* __restrict__ sel) {
  if (threadIdx.x != 0 || blockIdx.x != 0) return;
  unsigned int cum = 0;
  for (int i = 0; i < 65536; ++i) {
    unsigned int h = hist[i];
    if (cum + h > MEDIAN_RANK) { sel[0] = (unsigned int)i; sel[1] = MEDIAN_RANK - cum; return; }
    cum += h;
  }
  sel[0] = 65535u; sel[1] = 0u;
}

__global__ void k_hist_lo(const float* __restrict__ w, const unsigned int* __restrict__ sel,
                          unsigned int* __restrict__ hist) {
  unsigned int hibin = sel[0];
  unsigned int i = blockIdx.x * blockDim.x + threadIdx.x;
  unsigned int stride = gridDim.x * blockDim.x;
  for (unsigned int idx = i; idx < NW; idx += stride) {
    unsigned int b = __float_as_uint(fabsf(w[idx]));
    if ((b >> 16) == hibin) atomicAdd(&hist[b & 0xFFFFu], 1u);
  }
}

__global__ void k_sel_lo(const unsigned int* __restrict__ hist, unsigned int* __restrict__ sel) {
  if (threadIdx.x != 0 || blockIdx.x != 0) return;
  unsigned int rank = sel[1];
  unsigned int cum = 0;
  for (int i = 0; i < 65536; ++i) {
    unsigned int h = hist[i];
    if (cum + h > rank) { sel[2] = (sel[0] << 16) | (unsigned int)i; return; }
    cum += h;
  }
  sel[2] = (sel[0] << 16) | 0xFFFFu;
}

// ------------- ternarize W[n][k] -> bf16 WqT[k][n] (tiled transpose) -------------

__global__ __launch_bounds__(256) void k_quantT(const float* __restrict__ w,
                                                const unsigned int* __restrict__ sel,
                                                unsigned short* __restrict__ wqT) {
  __shared__ unsigned short tile[32][33];
  const float th = __uint_as_float(sel[2]);
  const int tx = threadIdx.x;          // 0..31
  const int ty = threadIdx.y;          // 0..7
  const int k0 = blockIdx.x * 32;
  const int n0 = blockIdx.y * 32;
#pragma unroll
  for (int j = 0; j < 4; ++j) {
    int nr = ty + j * 8;
    float v = w[(size_t)(n0 + nr) * KDIM + k0 + tx];   // coalesced along k
    unsigned short q = 0u;
    if (v > th)       q = 0x3F80u;   // +1.0 bf16
    else if (v < -th) q = 0xBF80u;   // -1.0 bf16
    tile[nr][tx] = q;
  }
  __syncthreads();
#pragma unroll
  for (int j = 0; j < 4; ++j) {
    int kr = ty + j * 8;
    wqT[(size_t)(k0 + kr) * NDIM + n0 + tx] = tile[tx][kr];  // coalesced along n
  }
}

// ------------------------------ bf16 WMMA GEMM ------------------------------
// Block tile 128(M) x 128(N), BK=64, double-buffered LDS.
// 256 threads = 8 waves in a 4(M) x 2(N) grid; each wave owns 2x4 16x16 f32 tiles.
// B tile staged via GLOBAL_LOAD_ASYNC_TO_LDS_B128 (ASYNCcnt) when available.

#define SA_STRIDE 72    // 64 + 8 halves pad
#define SB_STRIDE 136   // 128 + 8 halves pad
#define NT (KDIM / 64)

__global__ __launch_bounds__(256) void ternary_gemm(
    const float* __restrict__ x, const unsigned short* __restrict__ wqT,
    const float* __restrict__ log_scale, float* __restrict__ out) {
  __shared__ unsigned short sA[2][128 * SA_STRIDE];  // [m][k] bf16
  __shared__ unsigned short sB[2][64 * SB_STRIDE];   // [k][n] bf16

  const int tid  = threadIdx.x;
  const int lane = tid & 31;
  const int wid  = tid >> 5;
  const int wm   = wid & 3;     // 0..3 -> M offset wm*32
  const int wn   = wid >> 2;    // 0..1 -> N offset wn*64
  const int m0   = blockIdx.y * 128;
  const int n0   = blockIdx.x * 128;

  v8f acc[2][4];
#pragma unroll
  for (int mt = 0; mt < 2; ++mt)
#pragma unroll
    for (int nt = 0; nt < 4; ++nt)
#pragma unroll
      for (int v = 0; v < 8; ++v) acc[mt][nt][v] = 0.0f;

  const int lr = lane & 15;     // row-within-subtile / n-within-subtile
  const int lh = lane >> 4;     // lane-half selector

  // ---- stagers ----
  auto loadA = [&](int buf, int t0) {   // x f32 -> bf16, 128 x 64
#pragma unroll
    for (int i = 0; i < 8; ++i) {
      int idx = i * 256 + tid;          // float4 slots, 16 per row
      int row = idx >> 4;
      int c4  = idx & 15;
      const float4 f = *(const float4*)(x + (size_t)(m0 + row) * KDIM + t0 + c4 * 4);
      unsigned int lo = (unsigned int)f32_to_bf16_rne(f.x) |
                        ((unsigned int)f32_to_bf16_rne(f.y) << 16);
      unsigned int hi = (unsigned int)f32_to_bf16_rne(f.z) |
                        ((unsigned int)f32_to_bf16_rne(f.w) << 16);
      u32x2 p; p.x = lo; p.y = hi;
      *(u32x2*)(&sA[buf][row * SA_STRIDE + c4 * 4]) = p;
    }
  };
  auto loadB = [&](int buf, int t0) {   // WqT bf16 tile, 64 x 128
#pragma unroll
    for (int i = 0; i < 4; ++i) {
      int idx = i * 256 + tid;          // uint4 slots, 16 per row
      int row = idx >> 4;
      int c8  = idx & 15;
      const unsigned short* gp = wqT + (size_t)(t0 + row) * NDIM + n0 + c8 * 8;
      unsigned short* lp = &sB[buf][row * SB_STRIDE + c8 * 8];
#if HAVE_ASYNC_LDS
      __builtin_amdgcn_global_load_async_to_lds_b128(
          (as1_v4i_ptr)(void*)gp, (as3_v4i_ptr)(void*)lp, 0, 0);
#else
      *(u32x4*)lp = *(const u32x4*)gp;
#endif
    }
  };
  auto compute = [&](int buf) {
#pragma unroll
    for (int ks = 0; ks < 2; ++ks) {
      const int kk = ks * 32;
      union { v16bf v; u32x4 q[2]; } a[2], b[4];
      // A frag (16x32 bf16): lane m -> K 0..7 & 16..23 ; lane m+16 -> K 8..15 & 24..31
#pragma unroll
      for (int mt = 0; mt < 2; ++mt) {
        int row = wm * 32 + mt * 16 + lr;
        a[mt].q[0] = *(const u32x4*)(&sA[buf][row * SA_STRIDE + kk + lh * 8]);
        a[mt].q[1] = *(const u32x4*)(&sA[buf][row * SA_STRIDE + kk + 16 + lh * 8]);
      }
      // B frag (32x16 bf16): lane n -> K 0..7 & 16..23 ; lane n+16 -> K 8..15 & 24..31
#pragma unroll
      for (int nt = 0; nt < 4; ++nt) {
        int col = wn * 64 + nt * 16;
#pragma unroll
        for (int j = 0; j < 4; ++j) {
          int kb = kk + lh * 8 + 2 * j;
          b[nt].q[0][j] = (unsigned int)sB[buf][kb * SB_STRIDE + col + lr] |
                          ((unsigned int)sB[buf][(kb + 1) * SB_STRIDE + col + lr] << 16);
          b[nt].q[1][j] = (unsigned int)sB[buf][(kb + 16) * SB_STRIDE + col + lr] |
                          ((unsigned int)sB[buf][(kb + 17) * SB_STRIDE + col + lr] << 16);
        }
      }
#pragma unroll
      for (int mt = 0; mt < 2; ++mt)
#pragma unroll
        for (int nt = 0; nt < 4; ++nt)
          acc[mt][nt] = __builtin_amdgcn_wmma_f32_16x16x32_bf16(
              false, a[mt].v, false, b[nt].v, (short)0, acc[mt][nt], false, false);
    }
  };

  // ---- pipelined main loop: one barrier per K-tile ----
  loadA(0, 0);
  loadB(0, 0);
  WAIT_ASYNC0();
  __syncthreads();

#pragma unroll 2
  for (int t = 0; t < NT; ++t) {
    const int cur = t & 1;
    if (t + 1 < NT) {
      loadA(cur ^ 1, (t + 1) * 64);
      loadB(cur ^ 1, (t + 1) * 64);
    }
    if (t + 2 < NT) {
      // warm L2 for the A tile after next (global_prefetch_b8)
      __builtin_prefetch(x + (size_t)(m0 + (tid >> 1)) * KDIM + (t + 2) * 64 + (tid & 1) * 32, 0, 1);
    }
    compute(cur);
    WAIT_ASYNC0();
    __syncthreads();
  }

  // ---- epilogue: per-column softplus scale (exact, f32), then store ----
#pragma unroll
  for (int nt = 0; nt < 4; ++nt) {
    int n = n0 + wn * 64 + nt * 16 + lr;
    float ls = log_scale[n];
    float sc = fmaxf(ls, 0.0f) + log1pf(__expf(-fabsf(ls)));
#pragma unroll
    for (int mt = 0; mt < 2; ++mt) {
      int mbase = m0 + wm * 32 + mt * 16 + lh * 8;   // C/D: vgpr r -> M=r (lanes 0-15) / M=r+8
#pragma unroll
      for (int v = 0; v < 8; ++v) {
        out[(size_t)(mbase + v) * NDIM + n] = acc[mt][nt][v] * sc;
      }
    }
  }
}

// ---------------------------------- launcher ----------------------------------

extern "C" void kernel_launch(void* const* d_in, const int* in_sizes, int n_in,
                              void* d_out, int out_size, void* d_ws, size_t ws_size,
                              hipStream_t stream) {
  (void)in_sizes; (void)n_in; (void)out_size; (void)ws_size;
  const float* x  = (const float*)d_in[0];
  const float* w  = (const float*)d_in[1];
  const float* ls = (const float*)d_in[2];
  float* out = (float*)d_out;

  unsigned int*  hist1 = (unsigned int*)d_ws;                 // 65536 u32
  unsigned int*  hist2 = hist1 + 65536;                       // 65536 u32
  unsigned int*  sel   = hist1 + 131072;                      // 4 u32
  unsigned short* wqT  = (unsigned short*)((char*)d_ws + 528384);  // 2048*2048 bf16 (8 MB)

  k_zero<<<(131076 + 255) / 256, 256, 0, stream>>>(hist1, 131076);
  k_hist_hi<<<2048, 256, 0, stream>>>(w, hist1);
  k_sel_hi<<<1, 32, 0, stream>>>(hist1, sel);
  k_hist_lo<<<2048, 256, 0, stream>>>(w, sel, hist2);
  k_sel_lo<<<1, 32, 0, stream>>>(hist2, sel);
  k_quantT<<<dim3(KDIM / 32, NDIM / 32), dim3(32, 8), 0, stream>>>(w, sel, wqT);
  ternary_gemm<<<dim3(NDIM / 128, MDIM / 128), 256, 0, stream>>>(x, wqT, ls, out);
}
